// Network_73512660238715
// MI455X (gfx1250) — matrix operand order repaired and checked by reading, not compiled, not run on hardware
//
#include <hip/hip_runtime.h>
#include <math.h>

typedef __attribute__((ext_vector_type(16))) _Float16 v16h;
typedef __attribute__((ext_vector_type(8)))  float    v8f;

#define KDIM 128   // inner dimension of every GEMM in this network
#define TB   256

// ---------------------------------------------------------------- utilities
__global__ void k_fill1(float* __restrict__ p, int n) {
    int i = blockIdx.x * blockDim.x + threadIdx.x;
    if (i < n) p[i] = 1.0f;            // deg starts at 1 (self loop)
}

__global__ void k_deg_accum(const int* __restrict__ dst, float* __restrict__ deg, int e) {
    int i = blockIdx.x * blockDim.x + threadIdx.x;
    if (i < e) atomicAdd(&deg[dst[i]], 1.0f);
}

__global__ void k_rsqrt_inplace(float* __restrict__ p, int n) {
    int i = blockIdx.x * blockDim.x + threadIdx.x;
    if (i < n) p[i] = rsqrtf(p[i]);
}

__global__ void k_cvt_f16(const float* __restrict__ in, _Float16* __restrict__ out, int n) {
    int i = blockIdx.x * blockDim.x + threadIdx.x;
    if (i < n) out[i] = (_Float16)in[i];
}

// Pack W[K,O] (f32 row-major) into WMMA B-fragment order, converted to f16.
// Flat packed index = (((colTile*ksteps + ks)*32 + lane)*16 + h)
//   K   = ks*32 + (lane>=16 ? 16 : 0) + h     (this code's contiguous K order: lanes 0-15 K=0..15, 16-31 K=16..31)
//   col = colTile*16 + (lane & 15)
__global__ void k_pack_B(const float* __restrict__ W, _Float16* __restrict__ Bp, int O) {
    int i = blockIdx.x * blockDim.x + threadIdx.x;
    int total = KDIM * O;
    if (i >= total) return;
    int h    = i & 15;
    int lane = (i >> 4) & 31;
    int rest = i >> 9;                  // colTile*ksteps + ks
    const int ksteps = KDIM >> 5;       // 4
    int ct = rest / ksteps;
    int ks = rest - ct * ksteps;
    int K  = (ks << 5) + ((lane >> 4) << 4) + h;
    int nc = (ct << 4) + (lane & 15);
    Bp[i] = (_Float16)W[(size_t)K * O + nc];
}

// ---------------------------------------------------------------- fused WMMA GEMM
// One wave computes one 16-row strip across ALL CT column tiles:
//   - A fragment loaded once per K-step (2x b128), reused for CT WMMAs
//   - packed B staged in LDS once per workgroup (ds_load_b128 in inner loop)
//   - epilogue fuses the GCN self-loop term: agg = h*dinv^2 + bias
template <int CT>
__global__ void k_gemm_fused(const _Float16* __restrict__ A,
                             const _Float16* __restrict__ Bp,
                             const float* __restrict__ dinv,
                             const float* __restrict__ bias,
                             float* __restrict__ hbuf,
                             float* __restrict__ agg,
                             int nrows) {
    constexpr int O      = CT * 16;
    constexpr int KSTEPS = KDIM / 32;            // 4
    constexpr int NQ     = CT * KSTEPS * 32 * 2; // float4 count = KDIM*O/8
    __shared__ float4 ldsq[NQ];                  // 32KB for CT=8

    // cooperative stage of packed weights into LDS
    const int tid = threadIdx.x;
    const float4* gq = (const float4*)Bp;
    for (int i = tid; i < NQ; i += blockDim.x) ldsq[i] = gq[i];
    __syncthreads();

    const int lane    = tid & 31;
    const int rowTile = blockIdx.x * (blockDim.x >> 5) + (tid >> 5);
    if (rowTile >= (nrows >> 4)) return;
    const int hi  = lane >> 4;
    const int l15 = lane & 15;
    const int m   = (rowTile << 4) + l15;

    v8f acc[CT];
#pragma unroll
    for (int ct = 0; ct < CT; ++ct) acc[ct] = (v8f){};

#pragma unroll
    for (int ks = 0; ks < KSTEPS; ++ks) {
        // A fragment (16-bit A 16x32): halves 0..15 -> K = k0+hi*16+0..15
        union { v16h v; float4 q[2]; } au;
        const _Float16* ap = A + (size_t)m * KDIM + (ks << 5) + hi * 16;
        au.q[0] = *(const float4*)(ap);
        au.q[1] = *(const float4*)(ap + 8);
#pragma unroll
        for (int ct = 0; ct < CT; ++ct) {
            union { v16h v; float4 q[2]; } bu;
            const int idx = ((ct * KSTEPS + ks) * 32 + lane) * 2;
            bu.q[0] = ldsq[idx];
            bu.q[1] = ldsq[idx + 1];
            acc[ct] = __builtin_amdgcn_wmma_f32_16x16x32_f16(
                false, au.v, false, bu.v, (short)0, acc[ct], false, false);
        }
    }

    // bias per column tile (uniform over rows)
    float bv[CT];
#pragma unroll
    for (int ct = 0; ct < CT; ++ct) bv[ct] = bias[(ct << 4) + l15];

    // D layout: lanes 0-15 -> rows 0..7 in vgpr 0..7 ; lanes 16-31 -> rows 8..15
    const int crow0 = (rowTile << 4) + hi * 8;
#pragma unroll
    for (int j = 0; j < 8; ++j) {
        const float di  = dinv[crow0 + j];
        const float di2 = di * di;
        const size_t rb = (size_t)(crow0 + j) * O + l15;
#pragma unroll
        for (int ct = 0; ct < CT; ++ct) {
            const float h = acc[ct][j];
            hbuf[rb + (ct << 4)] = h;
            agg[rb + (ct << 4)] = h * di2 + bv[ct];
        }
    }
}

// ---------------------------------------------------------------- GCN aggregation
// agg[dst] += h[src] * dinv[src]*dinv[dst]
// 1 thread = 4 features of 1 edge; LOGC = log2(ncols/4) so the edge/chunk
// decomposition is shift+mask (no runtime divide in the hottest kernel).
template <int LOGC>
__global__ void k_edge_scatter(const float* __restrict__ h, const int* __restrict__ src,
                               const int* __restrict__ dst, const float* __restrict__ dinv,
                               float* __restrict__ agg, int e) {
    constexpr int CH = 1 << LOGC;   // float4 chunks per edge
    constexpr int NC = CH * 4;      // ncols
    int t = blockIdx.x * blockDim.x + threadIdx.x;
    int eidx = t >> LOGC;
    int c    = t & (CH - 1);
    if (eidx >= e) return;
    int s = src[eidx], d = dst[eidx];
    float coef = dinv[s] * dinv[d];
    float4 v = *(const float4*)(h + (size_t)s * NC + (c << 2));
    float* out = agg + (size_t)d * NC + (c << 2);
    atomicAdd(out + 0, v.x * coef);
    atomicAdd(out + 1, v.y * coef);
    atomicAdd(out + 2, v.z * coef);
    atomicAdd(out + 3, v.w * coef);
}

// activation in-place on agg (mode 0: id, 1: tanh, 2: leaky_relu 0.01),
// and emit f16 copy for the next layer's WMMA GEMM input (layers 1..3 only)
__global__ void k_act(float* __restrict__ agg, _Float16* __restrict__ xh,
                      int total, int mode) {
    int i = blockIdx.x * blockDim.x + threadIdx.x;
    if (i >= total) return;
    float v = agg[i];
    if (mode == 1)      v = tanhf(v);
    else if (mode == 2) v = (v > 0.0f) ? v : 0.01f * v;
    agg[i] = v;
    xh[i] = (_Float16)v;
}

// out[i] = elu(dot(leaky_relu(agg4[i,0:16]), Wc) + bc)   (layer-4 act fused in)
__global__ void k_final(const float* __restrict__ a4, const float* __restrict__ Wc,
                        const float* __restrict__ bc, float* __restrict__ out, int nrows) {
    int i = blockIdx.x * blockDim.x + threadIdx.x;
    if (i >= nrows) return;
    const float* row = a4 + (size_t)i * 16;
    float s = bc[0];
#pragma unroll
    for (int j = 0; j < 16; ++j) {
        float v = row[j];
        v = (v > 0.0f) ? v : 0.01f * v;     // leaky_relu from layer 4
        s += v * Wc[j];
    }
    out[i] = (s > 0.0f) ? s : (expf(s) - 1.0f);   // elu
}

// ---------------------------------------------------------------- driver
extern "C" void kernel_launch(void* const* d_in, const int* in_sizes, int n_in,
                              void* d_out, int out_size, void* d_ws, size_t ws_size,
                              hipStream_t stream) {
    const float* x  = (const float*)d_in[0];
    const int*   ei = (const int*)d_in[1];
    const float* W[4] = { (const float*)d_in[2], (const float*)d_in[4],
                          (const float*)d_in[6], (const float*)d_in[8] };
    const float* B[4] = { (const float*)d_in[3], (const float*)d_in[5],
                          (const float*)d_in[7], (const float*)d_in[9] };
    const float* Wc = (const float*)d_in[10];
    const float* bc = (const float*)d_in[11];

    const int F = 128, H = 128;
    const int N = in_sizes[0] / F;
    const int E = in_sizes[1] / 2;
    const int* src = ei;
    const int* dst = ei + E;
    const int outDim[4]  = { H, H, H, 16 };
    const int actMode[4] = { 0, 1, 2, 2 };

    // workspace carve (256B aligned slices)
    char*  ws  = (char*)d_ws;
    size_t off = 0;
    auto carve = [&](size_t bytes) -> void* {
        void* p = ws + off;
        off = (off + bytes + 255) & ~(size_t)255;
        return p;
    };
    _Float16* xh   = (_Float16*)carve((size_t)N * H * sizeof(_Float16));
    float*    hbuf = (float*)   carve((size_t)N * H * sizeof(float));
    float*    agg  = (float*)   carve((size_t)N * H * sizeof(float));
    float*    dinv = (float*)   carve((size_t)N * sizeof(float));
    _Float16* Wh   = (_Float16*)carve((size_t)KDIM * H * sizeof(_Float16));
    (void)ws_size; (void)n_in; (void)out_size;

    auto blocks = [](long long n) { return dim3((unsigned)((n + TB - 1) / TB)); };

    // degrees: deg = 1 + scatter-count(dst); dinv = rsqrt(deg)
    k_fill1         <<<blocks(N), TB, 0, stream>>>(dinv, N);
    k_deg_accum     <<<blocks(E), TB, 0, stream>>>(dst, dinv, E);
    k_rsqrt_inplace <<<blocks(N), TB, 0, stream>>>(dinv, N);

    // x -> f16 GEMM input
    k_cvt_f16<<<blocks((long long)N * F), TB, 0, stream>>>(x, xh, N * F);

    const int rowTiles = N >> 4;
    const int wpb      = TB / 32;
    const dim3 gemmGrid((rowTiles + wpb - 1) / wpb);

    for (int l = 0; l < 4; ++l) {
        const int O = outDim[l];
        // pack W[l] into WMMA B-fragment layout (f16)
        k_pack_B<<<blocks((long long)KDIM * O), TB, 0, stream>>>(W[l], Wh, O);
        // h = x @ W via v_wmma_f32_16x16x32_f16, fused with agg = h*dinv^2 + b
        if (O == 128) {
            k_gemm_fused<8><<<gemmGrid, TB, 0, stream>>>(xh, Wh, dinv, B[l], hbuf, agg, N);
            k_edge_scatter<5><<<blocks((long long)E * 32), TB, 0, stream>>>(hbuf, src, dst, dinv, agg, E);
        } else {
            k_gemm_fused<1><<<gemmGrid, TB, 0, stream>>>(xh, Wh, dinv, B[l], hbuf, agg, N);
            k_edge_scatter<2><<<blocks((long long)E * 4), TB, 0, stream>>>(hbuf, src, dst, dinv, agg, E);
        }
        // activation (+ f16 for next layer); layer-4 activation is fused into k_final
        if (l < 3)
            k_act<<<blocks((long long)N * O), TB, 0, stream>>>(agg, xh, N * O, actMode[l]);
    }

    // out = elu(leaky_relu(agg4) @ Wc + bc)
    k_final<<<blocks(N), TB, 0, stream>>>(agg, Wc, bc, (float*)d_out, N);
}
